// FlowNet3D_40415642255754
// MI455X (gfx1250) — matrix operand order repaired and checked by reading, not compiled, not run on hardware
//
#include <hip/hip_runtime.h>

// ---------------------------------------------------------------------------
// PointNet++-style double set-abstraction for two frames, CDNA5 (gfx1250).
//  - FPS: 1 workgroup/batch, whole cloud resident in LDS (320KB/WGP), min_d in registers
//  - Ball query: 1 wave32/query, ballot + prefix-popc ordered selection
//  - Grouping + MLP + maxpool fused: 1 wave32/group, MLP via
//    v_wmma_f32_16x16x32_f16 (K=16 group size == WMMA M dim), fp16 in/fp32 acc
//  - Weights pre-swizzled in LDS into fragment-ready layout so every
//    B-fragment is one contiguous 32B LDS load (2x ds_load_b128).
// ---------------------------------------------------------------------------

typedef _Float16 v16h __attribute__((ext_vector_type(16)));
typedef _Float16 v8h  __attribute__((ext_vector_type(8)));
typedef float    v8f  __attribute__((ext_vector_type(8)));

#define B_  8
#define N1_ 4096
#define S_  1024
#define K_  16

// ---------------------------- FPS ------------------------------------------
template<int PPT>
__global__ void fps_kernel(const float* __restrict__ xyz, int N, int S,
                           int* __restrict__ fidx, float* __restrict__ nxyz)
{
    __shared__ float sx[4096], sy[4096], sz[4096];
    __shared__ float rv[8];
    __shared__ int   ri[8];
    __shared__ int   sFar;

    const int b = blockIdx.x;
    const int t = threadIdx.x;
    const float* base = xyz + (size_t)b * N * 3;

    float mind[PPT];
#pragma unroll
    for (int j = 0; j < PPT; ++j) {
        int i = t + j * 256;
        if (i < N) {
            sx[i] = base[i * 3 + 0];
            sy[i] = base[i * 3 + 1];
            sz[i] = base[i * 3 + 2];
        }
        mind[j] = 1e9f;
    }
    __syncthreads();

    const int lane = t & 31, wid = t >> 5;
    int far = 0;
    for (int s = 0; s < S; ++s) {
        float fx = sx[far], fy = sy[far], fz = sz[far];
        if (t == 0) {
            fidx[b * S + s] = far;
            nxyz[(b * S + s) * 3 + 0] = fx;
            nxyz[(b * S + s) * 3 + 1] = fy;
            nxyz[(b * S + s) * 3 + 2] = fz;
        }
        float bv = -1.0f; int bi = 0x7fffffff;
#pragma unroll
        for (int j = 0; j < PPT; ++j) {
            int i = t + j * 256;
            if (i < N) {
                float dx = sx[i] - fx, dy = sy[i] - fy, dz = sz[i] - fz;
                float d  = dx * dx + dy * dy + dz * dz;
                float md = mind[j] < d ? mind[j] : d;
                mind[j]  = md;
                if (md > bv || (md == bv && i < bi)) { bv = md; bi = i; }
            }
        }
        // wave32 argmax reduction (prefer lower index on tie == jnp.argmax)
#pragma unroll
        for (int off = 16; off > 0; off >>= 1) {
            float ov = __shfl_down(bv, off, 32);
            int   oi = __shfl_down(bi, off, 32);
            if (ov > bv || (ov == bv && oi < bi)) { bv = ov; bi = oi; }
        }
        if (lane == 0) { rv[wid] = bv; ri[wid] = bi; }
        __syncthreads();
        if (t == 0) {
            float v = rv[0]; int i0 = ri[0];
#pragma unroll
            for (int w = 1; w < 8; ++w)
                if (rv[w] > v || (rv[w] == v && ri[w] < i0)) { v = rv[w]; i0 = ri[w]; }
            sFar = i0;
        }
        __syncthreads();
        far = sFar;
    }
}

// ------------------------- Ball query --------------------------------------
__global__ void ball_query_kernel(const float* __restrict__ q,
                                  const float* __restrict__ xyz,
                                  int N, float r2, int* __restrict__ out)
{
    __shared__ int sel[8][16];
    const int lane  = threadIdx.x & 31;
    const int wslot = threadIdx.x >> 5;
    const int w     = blockIdx.x * 8 + wslot;   // global query id = b*S + s
    const int b     = w / S_;

    const float qx = q[w * 3 + 0], qy = q[w * 3 + 1], qz = q[w * 3 + 2];
    const float* base = xyz + (size_t)b * N * 3;

    int cnt = 0;
    for (int bs = 0; bs < N && cnt < 16; bs += 32) {
        int i = bs + lane;
        float dx = base[i * 3 + 0] - qx;
        float dy = base[i * 3 + 1] - qy;
        float dz = base[i * 3 + 2] - qz;
        bool in = (dx * dx + dy * dy + dz * dz) <= r2;
        unsigned mask = (unsigned)__ballot(in);
        int r = cnt + __popc(mask & ((1u << lane) - 1u));
        if (in && r < 16) sel[wslot][r] = i;
        cnt += __popc(mask);
    }
    __syncthreads();   // intra-wave LDS visibility (uniform across block)
    if (lane < 16) {
        int v = (lane < cnt) ? sel[wslot][lane] : sel[wslot][0];
        out[(size_t)w * K_ + lane] = v;
    }
}

// ---------------------- WMMA fragment helpers -------------------------------
// A (16x32 f16): lanes 0-15 row M=lane hold K=8g..8g+7 (a[0..7]) and
// K=16+8g..16+8g+7 (a[8..15]) -> two contiguous 16B LDS loads per fragment.
__device__ __forceinline__ v16h a_frag(const _Float16* row, int ktile, int g)
{
    const v8h lo = *(const v8h*)(row + ktile * 32 + 8 * g);
    const v8h hi = *(const v8h*)(row + ktile * 32 + 16 + 8 * g);
    v16h a;
#pragma unroll
    for (int e = 0; e < 8; ++e) { a[e] = lo[e]; a[8 + e] = hi[e]; }
    return a;
}

// Swizzled weight fill: dst[((kt*NT+nt)*32 + lane)*16 + e] = W[k][o] (f16, padded)
// with k = kt*32 + (e<8 ? 8*(lane>>4)+e : 16+8*(lane>>4)+e-8), o = nt*16 + (lane&15). One-time cost,
// makes every B-fragment a single contiguous 32-byte LDS region.
__device__ __forceinline__ void load_wswz(_Float16* dst, const float* __restrict__ W,
                                          int Cin, int O, int KT, int tcount, int tid)
{
    const int NT = O >> 4;
    const int total = KT * NT * 512;
    for (int i = tid; i < total; i += tcount) {
        int e     = i & 15;
        int lane_ = (i >> 4) & 31;
        int tile  = i >> 9;
        int nt    = tile % NT, kt = tile / NT;
        int k     = kt * 32 + ((e < 8) ? 8 * (lane_ >> 4) + e : 16 + 8 * (lane_ >> 4) + (e - 8));
        int o     = nt * 16 + (lane_ & 15);
        dst[i] = (_Float16)((k < Cin) ? W[k * O + o] : 0.0f);
    }
}
__device__ __forceinline__ v16h b_frag(const _Float16* wswz, int NT,
                                       int kt, int nt, int lane)
{
    return *(const v16h*)(wswz + (((kt * NT) + nt) * 32 + lane) * 16);
}
__device__ __forceinline__ v8f bias_c(const float* bias, int ntile, int n)
{
    v8f c; float bv = bias[ntile * 16 + n];
#pragma unroll
    for (int r = 0; r < 8; ++r) c[r] = bv;
    return c;
}
// C layout: lane col n=lane&15, VGPR r -> row r + 8*(lane>>4). ReLU + restage.
__device__ __forceinline__ void store_relu(_Float16* st, int stride,
                                           const v8f& c, int ntile, int n, int g)
{
#pragma unroll
    for (int r = 0; r < 8; ++r) {
        float x = c[r]; x = x > 0.0f ? x : 0.0f;
        st[(r + 8 * g) * stride + ntile * 16 + n] = (_Float16)x;
    }
}
#define WMMA_F16(A, Bf, C) \
    __builtin_amdgcn_wmma_f32_16x16x32_f16(false, (A), false, (Bf), (short)0, (C), false, false)

// ---------------------- SA level 1: 3(->32 pad) -> 32 -> 32 -> 64 ----------
__global__ void sa1_kernel(const float* __restrict__ xyz,
                           const float* __restrict__ nxyz,
                           const int*   __restrict__ idx,
                           const float* __restrict__ W1, const float* __restrict__ bb1,
                           const float* __restrict__ W2, const float* __restrict__ bb2,
                           const float* __restrict__ W3, const float* __restrict__ bb3,
                           float* __restrict__ outp)
{
    __shared__ __align__(32) _Float16 w1p[1 * 2 * 512];   // KT=1 NT=2
    __shared__ __align__(32) _Float16 w2p[1 * 2 * 512];   // KT=1 NT=2
    __shared__ __align__(32) _Float16 w3p[1 * 4 * 512];   // KT=1 NT=4
    __shared__ float b1s[32], b2s[32], b3s[64];
    __shared__ __align__(32) _Float16 stage[8][16 * 32];

    const int t = threadIdx.x;
    load_wswz(w1p, W1, 3, 32, 1, 256, t);
    load_wswz(w2p, W2, 32, 32, 1, 256, t);
    load_wswz(w3p, W3, 32, 64, 1, 256, t);
    if (t < 32) { b1s[t] = bb1[t]; b2s[t] = bb2[t]; }
    if (t < 64) b3s[t] = bb3[t];
    __syncthreads();

    const int lane = t & 31, wslot = t >> 5;
    const int w    = blockIdx.x * 8 + wslot;       // group id = b*S + s
    const int bN   = w / S_;
    const int row  = lane & 15, g = lane >> 4, n = row;
    _Float16* st = &stage[wslot][0];

    // gather: feat row = grouped_xyz - center, padded to 32 ch, vector stores
    {
        int j = idx[(size_t)w * K_ + row];
        const float* p = xyz  + ((size_t)bN * N1_ + j) * 3;
        const float* c = nxyz + (size_t)w * 3;
        v8h lo, hi;
#pragma unroll
        for (int e = 0; e < 8; ++e) { lo[e] = (_Float16)0.0f; hi[e] = (_Float16)0.0f; }
        if (g == 0) {
            lo[0] = (_Float16)(p[0] - c[0]);
            lo[1] = (_Float16)(p[1] - c[1]);
            lo[2] = (_Float16)(p[2] - c[2]);
        }
        _Float16* r = st + row * 32 + g * 16;
        *(v8h*)(r)     = lo;
        *(v8h*)(r + 8) = hi;
    }
    // layer 1: 32 -> 32  (LDS ops within a wave are in-order: no barrier)
    {
        v16h a  = a_frag(st + row * 32, 0, g);
        v8f  c0 = bias_c(b1s, 0, n), c1 = bias_c(b1s, 1, n);
        c0 = WMMA_F16(a, b_frag(w1p, 2, 0, 0, lane), c0);
        c1 = WMMA_F16(a, b_frag(w1p, 2, 0, 1, lane), c1);
        store_relu(st, 32, c0, 0, n, g);
        store_relu(st, 32, c1, 1, n, g);
    }
    // layer 2: 32 -> 32
    {
        v16h a  = a_frag(st + row * 32, 0, g);
        v8f  c0 = bias_c(b2s, 0, n), c1 = bias_c(b2s, 1, n);
        c0 = WMMA_F16(a, b_frag(w2p, 2, 0, 0, lane), c0);
        c1 = WMMA_F16(a, b_frag(w2p, 2, 0, 1, lane), c1);
        store_relu(st, 32, c0, 0, n, g);
        store_relu(st, 32, c1, 1, n, g);
    }
    // layer 3: 32 -> 64, fused max over K=16 rows straight out of C fragments
    {
        v16h a = a_frag(st + row * 32, 0, g);
#pragma unroll
        for (int nt = 0; nt < 4; ++nt) {
            v8f c = bias_c(b3s, nt, n);
            c = WMMA_F16(a, b_frag(w3p, 4, 0, nt, lane), c);
            float m = c[0];
#pragma unroll
            for (int r = 1; r < 8; ++r) m = fmaxf(m, c[r]);
            m = fmaxf(m, __shfl_xor(m, 16, 32));   // other half of the rows
            m = fmaxf(m, 0.0f);                     // relu(max) == max(relu)
            if (lane < 16) outp[(size_t)w * 64 + nt * 16 + lane] = m;
        }
    }
}

// ------------- SA level 2: 67(->96 pad) -> 64 -> 64 -> 128 ------------------
__global__ void sa2_kernel(const float* __restrict__ l1xyz,   // [B][S][3]
                           const float* __restrict__ l2xyz,   // [B][S][3] centers
                           const int*   __restrict__ idx,     // [B][S][K]
                           const float* __restrict__ l1pts,   // [B][S][64]
                           const float* __restrict__ W1, const float* __restrict__ bb1,
                           const float* __restrict__ W2, const float* __restrict__ bb2,
                           const float* __restrict__ W3, const float* __restrict__ bb3,
                           float* __restrict__ outp)          // [B][S][128]
{
    __shared__ __align__(32) _Float16 w1p[3 * 4 * 512];   // KT=3 NT=4 (96x64)
    __shared__ __align__(32) _Float16 w2p[2 * 4 * 512];   // KT=2 NT=4 (64x64)
    __shared__ __align__(32) _Float16 w3p[2 * 8 * 512];   // KT=2 NT=8 (64x128)
    __shared__ float b1s[64], b2s[64], b3s[128];
    __shared__ __align__(32) _Float16 stage[4][16 * 96];

    const int t = threadIdx.x;
    load_wswz(w1p, W1, 67, 64, 3, 128, t);
    load_wswz(w2p, W2, 64, 64, 2, 128, t);
    load_wswz(w3p, W3, 64, 128, 2, 128, t);
    if (t < 64)  { b1s[t] = bb1[t]; b2s[t] = bb2[t]; }
    if (t < 128) b3s[t] = bb3[t];
    __syncthreads();

    const int lane = t & 31, wslot = t >> 5;
    const int w    = blockIdx.x * 4 + wslot;
    const int bN   = w / S_;
    const int row  = lane & 15, g = lane >> 4, n = row;
    _Float16* st = &stage[wslot][0];

    // gather: [diff(3) | l1pts(64) | pad(29)] = 96 ch; lane half g covers
    // channels [48g, 48g+48) as six 16-byte vector stores.
    {
        int j = idx[(size_t)w * K_ + row];
        const float* p  = l1xyz + ((size_t)bN * S_ + j) * 3;
        const float* cc = l2xyz + (size_t)w * 3;
        const float* f  = l1pts + ((size_t)bN * S_ + j) * 64;
        _Float16* r = st + row * 96 + g * 48;
        if (g == 0) {
            float dx = p[0] - cc[0], dy = p[1] - cc[1], dz = p[2] - cc[2];
#pragma unroll
            for (int chunk = 0; chunk < 6; ++chunk) {
                v8h vv;
#pragma unroll
                for (int e = 0; e < 8; ++e) {
                    int c = chunk * 8 + e;
                    float x = (c == 0) ? dx : (c == 1) ? dy : (c == 2) ? dz : f[c - 3];
                    vv[e] = (_Float16)x;
                }
                *(v8h*)(r + chunk * 8) = vv;
            }
        } else {
#pragma unroll
            for (int chunk = 0; chunk < 6; ++chunk) {
                v8h vv;
#pragma unroll
                for (int e = 0; e < 8; ++e) {
                    int c = 48 + chunk * 8 + e;
                    float x = (c < 67) ? f[c - 3] : 0.0f;
                    vv[e] = (_Float16)x;
                }
                *(v8h*)(r + chunk * 8) = vv;
            }
        }
    }
    // L1: 96 -> 64  (hoist all A reads before stage overwrite)
    {
        v16h a0 = a_frag(st + row * 96, 0, g);
        v16h a1 = a_frag(st + row * 96, 1, g);
        v16h a2 = a_frag(st + row * 96, 2, g);
#pragma unroll
        for (int nt = 0; nt < 4; ++nt) {
            v8f c = bias_c(b1s, nt, n);
            c = WMMA_F16(a0, b_frag(w1p, 4, 0, nt, lane), c);
            c = WMMA_F16(a1, b_frag(w1p, 4, 1, nt, lane), c);
            c = WMMA_F16(a2, b_frag(w1p, 4, 2, nt, lane), c);
            store_relu(st, 96, c, nt, n, g);
        }
    }
    // L2: 64 -> 64
    {
        v16h a0 = a_frag(st + row * 96, 0, g);
        v16h a1 = a_frag(st + row * 96, 1, g);
#pragma unroll
        for (int nt = 0; nt < 4; ++nt) {
            v8f c = bias_c(b2s, nt, n);
            c = WMMA_F16(a0, b_frag(w2p, 4, 0, nt, lane), c);
            c = WMMA_F16(a1, b_frag(w2p, 4, 1, nt, lane), c);
            store_relu(st, 96, c, nt, n, g);
        }
    }
    // L3: 64 -> 128 + fused maxpool over K=16
    {
        v16h a0 = a_frag(st + row * 96, 0, g);
        v16h a1 = a_frag(st + row * 96, 1, g);
#pragma unroll
        for (int nt = 0; nt < 8; ++nt) {
            v8f c = bias_c(b3s, nt, n);
            c = WMMA_F16(a0, b_frag(w3p, 8, 0, nt, lane), c);
            c = WMMA_F16(a1, b_frag(w3p, 8, 1, nt, lane), c);
            float m = c[0];
#pragma unroll
            for (int r = 1; r < 8; ++r) m = fmaxf(m, c[r]);
            m = fmaxf(m, __shfl_xor(m, 16, 32));
            m = fmaxf(m, 0.0f);
            if (lane < 16) outp[(size_t)w * 128 + nt * 16 + lane] = m;
        }
    }
}

// ---------------------------------------------------------------------------
extern "C" void kernel_launch(void* const* d_in, const int* in_sizes, int n_in,
                              void* d_out, int out_size, void* d_ws, size_t ws_size,
                              hipStream_t stream)
{
    (void)in_sizes; (void)n_in; (void)out_size; (void)ws_size;

    char* ws = (char*)d_ws;
    size_t off = 0;
    auto alloc = [&](size_t bytes) -> void* {
        void* p = ws + off;
        off = (off + bytes + 255) & ~(size_t)255;
        return p;
    };
    int*   fidx1 = (int*)  alloc((size_t)B_ * S_ * sizeof(int));
    float* nxyz1 = (float*)alloc((size_t)B_ * S_ * 3 * sizeof(float));
    int*   idx1  = (int*)  alloc((size_t)B_ * S_ * K_ * sizeof(int));
    float* l1pts = (float*)alloc((size_t)B_ * S_ * 64 * sizeof(float));
    int*   fidx2 = (int*)  alloc((size_t)B_ * S_ * sizeof(int));
    int*   idx2  = (int*)  alloc((size_t)B_ * S_ * K_ * sizeof(int));

    for (int f = 0; f < 2; ++f) {
        const float* xyz = (const float*)d_in[f];
        const int p0 = 2 + f * 12;
        const float* W11 = (const float*)d_in[p0 + 0];  const float* b11 = (const float*)d_in[p0 + 1];
        const float* W12 = (const float*)d_in[p0 + 2];  const float* b12 = (const float*)d_in[p0 + 3];
        const float* W13 = (const float*)d_in[p0 + 4];  const float* b13 = (const float*)d_in[p0 + 5];
        const float* W21 = (const float*)d_in[p0 + 6];  const float* b21 = (const float*)d_in[p0 + 7];
        const float* W22 = (const float*)d_in[p0 + 8];  const float* b22 = (const float*)d_in[p0 + 9];
        const float* W23 = (const float*)d_in[p0 + 10]; const float* b23 = (const float*)d_in[p0 + 11];

        float* l2xyz = (float*)d_out + (size_t)f * ((size_t)B_ * S_ * 3 + (size_t)B_ * S_ * 128);
        float* l2pts = l2xyz + (size_t)B_ * S_ * 3;

        // level 1
        fps_kernel<16><<<B_, 256, 0, stream>>>(xyz, N1_, S_, fidx1, nxyz1);
        ball_query_kernel<<<B_ * S_ / 8, 256, 0, stream>>>(nxyz1, xyz, N1_, 0.25f, idx1);
        sa1_kernel<<<B_ * S_ / 8, 256, 0, stream>>>(xyz, nxyz1, idx1,
                                                    W11, b11, W12, b12, W13, b13, l1pts);
        // level 2 (FPS over 1024 -> 1024 == ordering permutation; centers go to d_out)
        fps_kernel<4><<<B_, 256, 0, stream>>>(nxyz1, S_, S_, fidx2, l2xyz);
        ball_query_kernel<<<B_ * S_ / 8, 256, 0, stream>>>(l2xyz, nxyz1, S_, 1.0f, idx2);
        sa2_kernel<<<B_ * S_ / 4, 128, 0, stream>>>(nxyz1, l2xyz, idx2, l1pts,
                                                    W21, b21, W22, b22, W23, b23, l2pts);
    }
}
